// RNN_Hierarchical_12824772346217
// MI455X (gfx1250) — compile-verified, matrix operand order checked by reading, not run on hardware
//
#include <hip/hip_runtime.h>

typedef __attribute__((ext_vector_type(16))) _Float16 v16h;
typedef __attribute__((ext_vector_type(8)))  _Float16 v8h;
typedef __attribute__((ext_vector_type(8)))  float    v8f;

namespace {
constexpr int kT     = 512;   // timesteps
constexpr int kB     = 128;   // batch
constexpr int kD     = 3;     // modules
constexpr int kN     = 512;   // hidden
constexpr int kC     = 2;     // readout classes
constexpr int kNBlk  = 16;    // persistent workgroups (each on its own WGP)
constexpr int kThr   = 512;   // 16 wave32s per workgroup
constexpr int kSlice = kN / kNBlk;   // 32 output columns per block
constexpr int kHN    = kB * kN;      // elements per module state

constexpr size_t kSmemBytes =
    (size_t)(kD * kSlice * kN + (kD - 1) * kSlice * kN) * sizeof(_Float16)  // 160 KB weights
    + (size_t)4 * kD * kSlice * sizeof(float);                              // alpha/invtau/bias/win
}

// ---------------- device-wide barrier (sense via generation counter) --------
__device__ __forceinline__ void grid_barrier(unsigned* cnt, unsigned* gen,
                                             unsigned& local_gen) {
  __threadfence();          // flush this thread's stores to device scope
  __syncthreads();
  if (threadIdx.x == 0) {
    const unsigned target = local_gen + 1u;
    unsigned prev = __hip_atomic_fetch_add(cnt, 1u, __ATOMIC_ACQ_REL,
                                           __HIP_MEMORY_SCOPE_AGENT);
    if (prev == (unsigned)(kNBlk - 1)) {
      __hip_atomic_store(cnt, 0u, __ATOMIC_RELAXED, __HIP_MEMORY_SCOPE_AGENT);
      __hip_atomic_fetch_add(gen, 1u, __ATOMIC_RELEASE, __HIP_MEMORY_SCOPE_AGENT);
    } else {
      while (__hip_atomic_load(gen, __ATOMIC_ACQUIRE,
                               __HIP_MEMORY_SCOPE_AGENT) != target) {
        __builtin_amdgcn_s_sleep(2);
      }
    }
  }
  __syncthreads();
  __threadfence();          // acquire: invalidate WGP caches for all waves
  ++local_gen;
}

// One 16x16x32 f16 WMMA step. aptr/bptr already include +lhi*8 lane offset;
// the two halves of the fragment are K [0,8) and K [16,24) relative to kk.
__device__ __forceinline__ v8f wmma_step(const _Float16* __restrict__ aptr,
                                         const _Float16* __restrict__ bptr,
                                         v8f acc) {
  v8h alo = *(const v8h*)(aptr);
  v8h ahi = *(const v8h*)(aptr + 16);
  v8h blo = *(const v8h*)(bptr);
  v8h bhi = *(const v8h*)(bptr + 16);
  v16h a = __builtin_shufflevector(alo, ahi, 0,1,2,3,4,5,6,7,8,9,10,11,12,13,14,15);
  v16h b = __builtin_shufflevector(blo, bhi, 0,1,2,3,4,5,6,7,8,9,10,11,12,13,14,15);
  return __builtin_amdgcn_wmma_f32_16x16x32_f16(
      /*neg_a=*/false, a, /*neg_b=*/false, b,
      /*c_mod=*/(short)0, acc, /*reuse_a=*/false, /*reuse_b=*/false);
}

__global__ void rnn_init_sync(unsigned* sync) {
  if (threadIdx.x < 2) sync[threadIdx.x] = 0u;
}

__global__ void __launch_bounds__(kThr, 1)
rnn_hier_persistent(const float* __restrict__ data,   // [T,B,1]
                    const float* __restrict__ h0,     // [D,B,N]
                    const float* __restrict__ W_in,   // [D,N,1]
                    const float* __restrict__ b_in,   // [D,N]
                    const float* __restrict__ W_hh,   // [D,N,N]
                    const float* __restrict__ b_hh,   // [D,N]
                    const float* __restrict__ W_ff,   // [D-1,N,N]
                    const float* __restrict__ b_ff,   // [D-1,N]
                    const float* __restrict__ taus,   // [D,N]
                    const float* __restrict__ W_fc,   // [D,C,N]
                    const float* __restrict__ b_fc,   // [D,C]
                    float* __restrict__ out,          // h_final [D,B,N] ++ out [D,B,C]
                    _Float16* __restrict__ h16,       // [2][D][B][N] ping-pong
                    unsigned* __restrict__ sync) {
  extern __shared__ char smem[];
  _Float16* WhhL = (_Float16*)smem;                      // [D][kSlice][N]
  _Float16* WffL = WhhL + kD * kSlice * kN;              // [D-1][kSlice][N]
  float* alphaL  = (float*)(WffL + (kD - 1) * kSlice * kN);
  float* invtL   = alphaL + kD * kSlice;
  float* biasL   = invtL  + kD * kSlice;
  float* winL    = biasL  + kD * kSlice;

  const int tid = threadIdx.x;
  const int bx  = blockIdx.x;
  unsigned bar_gen = 0;

  // ---------------- prologue: stage f16 weight slices in LDS ---------------
  for (int idx = tid; idx < kD * kSlice * kN; idx += kThr) {
    int d  = idx / (kSlice * kN);
    int r  = idx % (kSlice * kN);
    int nl = r / kN, k = r % kN;
    int ng = bx * kSlice + nl;
    float w = W_hh[(d * kN + ng) * kN + k];
    if (ng == k) w = 0.0f;                       // zero recurrent diagonal
    WhhL[idx] = (_Float16)w;
  }
  for (int idx = tid; idx < (kD - 1) * kSlice * kN; idx += kThr) {
    int e  = idx / (kSlice * kN);
    int r  = idx % (kSlice * kN);
    int nl = r / kN, k = r % kN;
    int ng = bx * kSlice + nl;
    WffL[idx] = (_Float16)W_ff[(e * kN + ng) * kN + k];
  }
  for (int idx = tid; idx < kD * kSlice; idx += kThr) {
    int d = idx / kSlice, nl = idx % kSlice;
    int ng = bx * kSlice + nl;
    float tc = fmaxf(taus[d * kN + ng], 1.0f);
    alphaL[idx] = 1.0f - 1.0f / tc;
    invtL[idx]  = 1.0f / tc;
    float bias = b_hh[d * kN + ng] + b_in[d * kN + ng];
    if (d > 0) bias += b_ff[(d - 1) * kN + ng];
    biasL[idx] = bias;
    winL[idx]  = W_in[d * kN + ng];              // I == 1
  }
  // fp32 master state lives directly in d_out; f16 shadow (buffer 0) for WMMA
  float* hf32 = out;
  for (int idx = bx * kThr + tid; idx < kD * kHN; idx += kNBlk * kThr) {
    float v = h0[idx];
    hf32[idx] = v;
    h16[idx]  = (_Float16)v;
  }
  grid_barrier(sync, sync + 1, bar_gen);

  // ---------------- wave -> tile mapping -----------------------------------
  const int lane   = tid & 31;
  const int wv     = tid >> 5;          // 0..15
  const int m_base = (wv & 7) * 16;     // batch-tile
  const int n_locb = (wv >> 3) * 16;    // column-tile within slice
  const int llo    = lane & 15;
  const int lhi    = lane >> 4;
  const int mA     = m_base + llo;                 // A row (batch) this lane loads
  const int nB     = n_locb + llo;                 // weight row (local n) this lane loads
  const int nCg    = bx * kSlice + n_locb + llo;   // global column this lane updates

  // ---------------- time loop ----------------------------------------------
  for (int t = 0; t < kT; ++t) {
    const int p = t & 1;
    _Float16* hread  = h16 + p * kD * kHN;         // states entering step t
    _Float16* hwrite = h16 + (p ^ 1) * kD * kHN;   // states produced at step t
    const float* data_t = data + t * kB;           // [B]

    for (int d = 0; d < kD; ++d) {
      v8f acc = {0.f, 0.f, 0.f, 0.f, 0.f, 0.f, 0.f, 0.f};

      // recurrent: pre += h_old[d] @ Whh[d]^T
      const _Float16* Arow = hread + d * kHN + mA * kN + lhi * 8;
      const _Float16* Brow = WhhL + (d * kSlice + nB) * kN + lhi * 8;
      #pragma unroll 4
      for (int kk = 0; kk < kN; kk += 32)
        acc = wmma_step(Arow + kk, Brow + kk, acc);

      // feedforward: pre += h_new[d-1] @ Wff[d-1]^T
      if (d > 0) {
        const _Float16* Afr = hwrite + (d - 1) * kHN + mA * kN + lhi * 8;
        const _Float16* Bfr = WffL + ((d - 1) * kSlice + nB) * kN + lhi * 8;
        #pragma unroll 4
        for (int kk = 0; kk < kN; kk += 32)
          acc = wmma_step(Afr + kk, Bfr + kk, acc);
      }

      // pointwise leaky-integrator update (one lane owns each (b, n) element)
      const int pidx  = d * kSlice + n_locb + llo;
      const float alp = alphaL[pidx];
      const float ivt = invtL[pidx];
      const float bia = biasL[pidx];
      const float win = winL[pidx];
      float*     hS = hf32   + d * kHN;
      _Float16*  hW = hwrite + d * kHN;
      #pragma unroll
      for (int r = 0; r < 8; ++r) {
        const int m = m_base + r + 8 * lhi;        // batch index
        float pre = acc[r] + bia + data_t[m] * win;
        float act = pre > 0.0f ? pre : 0.01f * pre;   // leaky_relu
        float hn  = alp * hS[m * kN + nCg] + act * ivt;
        hS[m * kN + nCg] = hn;                     // fp32 master (== d_out)
        hW[m * kN + nCg] = (_Float16)hn;           // f16 shadow for next WMMA
      }
      grid_barrier(sync, sync + 1, bar_gen);       // publish module d's state
    }
  }

  // ---------------- epilogue: readout heads out[d,b,c] ---------------------
  float* out2 = out + kD * kHN;
  for (int o = bx * kThr + tid; o < kD * kB * kC; o += kNBlk * kThr) {
    int d = o / (kB * kC);
    int r = o % (kB * kC);
    int b = r / kC, c = r % kC;
    const float* hv = hf32 + (d * kB + b) * kN;
    const float* wv2 = W_fc + (d * kC + c) * kN;
    float s = b_fc[d * kC + c];
    for (int n = 0; n < kN; ++n) s += hv[n] * wv2[n];
    out2[o] = s;
  }
}

extern "C" void kernel_launch(void* const* d_in, const int* in_sizes, int n_in,
                              void* d_out, int out_size, void* d_ws, size_t ws_size,
                              hipStream_t stream) {
  const float* data = (const float*)d_in[0];
  const float* h0   = (const float*)d_in[1];
  const float* W_in = (const float*)d_in[2];
  const float* b_in = (const float*)d_in[3];
  const float* W_hh = (const float*)d_in[4];
  const float* b_hh = (const float*)d_in[5];
  const float* W_ff = (const float*)d_in[6];
  const float* b_ff = (const float*)d_in[7];
  const float* taus = (const float*)d_in[8];
  const float* W_fc = (const float*)d_in[9];
  const float* b_fc = (const float*)d_in[10];

  float*    out  = (float*)d_out;
  unsigned* sync = (unsigned*)d_ws;                       // 2 counters
  _Float16* h16  = (_Float16*)((char*)d_ws + 256);        // [2][D][B][N] f16

  // Allow >64KB dynamic LDS (WGP has 320KB). Idempotent, not a stream op.
  (void)hipFuncSetAttribute((const void*)rnn_hier_persistent,
                            hipFuncAttributeMaxDynamicSharedMemorySize,
                            (int)kSmemBytes);

  rnn_init_sync<<<1, 64, 0, stream>>>(sync);
  rnn_hier_persistent<<<kNBlk, kThr, kSmemBytes, stream>>>(
      data, h0, W_in, b_in, W_hh, b_hh, W_ff, b_ff, taus, W_fc, b_fc,
      out, h16, sync);
}
